// LlamaAttention_25159918420500
// MI455X (gfx1250) — matrix operand order repaired and checked by reading, not compiled, not run on hardware
//
#include <hip/hip_runtime.h>
#include <hip/hip_bf16.h>

// ---------------------------------------------------------------------------
// LlamaAttention for MI455X (gfx1250): bf16 WMMA everywhere, async-LDS
// double-buffered GEMM staging, prefetch in the attention stream.
// B=1, S=4096, HID=2048, H=32, KVH=8, D=64, N_REP=4, no causal mask.
// ---------------------------------------------------------------------------

typedef __bf16 bf16;
typedef __attribute__((ext_vector_type(16))) __bf16 v16bf;
typedef __attribute__((ext_vector_type(8)))  __bf16 v8bf;
typedef __attribute__((ext_vector_type(8)))  float  v8f;
typedef __attribute__((ext_vector_type(4)))  int    v4i;

union AFrag { v16bf v; v8bf h[2]; };

static __device__ inline v8f wmma_bf16(v16bf a, v16bf b, v8f c) {
  // D = A(16x32 bf16) * B(32x16 bf16) + C(16x16 f32)
  return __builtin_amdgcn_wmma_f32_16x16x32_bf16(
      /*neg_a=*/false, a, /*neg_b=*/false, b,
      /*c_mod=*/(short)0, c, /*reuse_a=*/false, /*reuse_b=*/false);
}

// ----- CDNA5 async global->LDS copy (ASYNCcnt-tracked), with fallback ------
#if defined(__has_builtin)
#if __has_builtin(__builtin_amdgcn_global_load_async_to_lds_b128) && \
    __has_builtin(__builtin_amdgcn_s_wait_asynccnt)
#define USE_ASYNC_COPY 1
#endif
#endif
#ifndef USE_ASYNC_COPY
#define USE_ASYNC_COPY 0
#endif

// copy 16 bytes (one b128) global -> LDS
static __device__ inline void copy_b128_to_lds(const bf16* g, bf16* l) {
#if USE_ASYNC_COPY
  __builtin_amdgcn_global_load_async_to_lds_b128(
      (__attribute__((address_space(1))) v4i*)g,
      (__attribute__((address_space(3))) v4i*)l,
      /*offset=*/0, /*cpol=*/0);
#else
  *(v8bf*)l = *(const v8bf*)g;
#endif
}

static __device__ inline void wait_async_copies() {
#if USE_ASYNC_COPY
  __builtin_amdgcn_s_wait_asynccnt(0);
#endif
}

// ---------------------------------------------------------------------------
// fp32 -> bf16 elementwise cast
// ---------------------------------------------------------------------------
__global__ void cast_f32_bf16(const float* __restrict__ in, bf16* __restrict__ out, int n) {
  int i = blockIdx.x * 256 + threadIdx.x;
  if (i < n) out[i] = (bf16)in[i];
}

// ---------------------------------------------------------------------------
// Generic bf16 GEMM:  C(MxN, f32) = A(MxK, bf16 row-major) * Bt(NxK, bf16)^T
// Bt is "weight layout" [out][in]  ==  B^T, matching the WMMA B-fragment
// (contiguous K per output column).
// Workgroup: 256 threads = 8 waves, tile 128(M) x 128(N), K-step 32,
// double-buffered LDS staged with async global->LDS copies.
// Wave (wm, wn) in 2x4 grid computes 64x32 via 4x2 accumulators.
// ---------------------------------------------------------------------------
#define GT_M 128
#define GT_N 128
#define GT_K 32

__global__ __launch_bounds__(256)
void gemm_bf16(const bf16* __restrict__ A, const bf16* __restrict__ Bt,
               float* __restrict__ C, int M, int N, int K) {
  __shared__ bf16 lA[2][GT_M * GT_K];   // [buf][m][k]
  __shared__ bf16 lB[2][GT_N * GT_K];   // [buf][n][k]  (B^T tile)

  const int tid  = threadIdx.x;
  const int lane = tid & 31;
  const int wave = tid >> 5;
  const int l16  = lane & 15;
  const int half = lane >> 4;
  const int wm   = wave >> 2;   // 0..1
  const int wn   = wave & 3;    // 0..3
  const int m0   = blockIdx.x * GT_M;
  const int n0   = blockIdx.y * GT_N;

  v8f acc[4][2];
#pragma unroll
  for (int i = 0; i < 4; ++i)
#pragma unroll
    for (int j = 0; j < 2; ++j) acc[i][j] = (v8f){};

  const int ldr = tid >> 1;          // 0..127 : tile row to load
  const int ldc = (tid & 1) * 16;    // 0 or 16 : k-chunk (16 bf16 = 32 B)

  auto stage = [&](int buf, int k0) {
    const bf16* ga = &A[(size_t)(m0 + ldr) * K + k0 + ldc];
    const bf16* gb = &Bt[(size_t)(n0 + ldr) * K + k0 + ldc];
    bf16* la = &lA[buf][ldr * GT_K + ldc];
    bf16* lb = &lB[buf][ldr * GT_K + ldc];
    copy_b128_to_lds(ga, la);
    copy_b128_to_lds(ga + 8, la + 8);
    copy_b128_to_lds(gb, lb);
    copy_b128_to_lds(gb + 8, lb + 8);
  };

  stage(0, 0);
  int buf = 0;

  for (int k0 = 0; k0 < K; k0 += GT_K) {
    wait_async_copies();
    __syncthreads();                         // tiles[buf] visible to all waves
    if (k0 + GT_K < K) stage(buf ^ 1, k0 + GT_K);

    // A fragments (16x32): lane l16 = row, element e -> k = half*16+e.
    v16bf aF[4];
#pragma unroll
    for (int mi = 0; mi < 4; ++mi) {
      AFrag f;
      const int r = wm * 64 + mi * 16 + l16;
      f.h[0] = *(const v8bf*)&lA[buf][r * GT_K + half * 16];
      f.h[1] = *(const v8bf*)&lA[buf][r * GT_K + half * 16 + 8];
      aF[mi] = f.v;
    }
    // B fragments (32x16): lane l16 = column, element e -> k = half*16+e.
    v16bf bF[2];
#pragma unroll
    for (int ni = 0; ni < 2; ++ni) {
      const int c = wn * 32 + ni * 16 + l16;
      bF[ni] = *(const v16bf*)&lB[buf][c * GT_K + half * 16];
    }
#pragma unroll
    for (int mi = 0; mi < 4; ++mi)
#pragma unroll
      for (int ni = 0; ni < 2; ++ni)
        acc[mi][ni] = wmma_bf16(aF[mi], bF[ni], acc[mi][ni]);

    buf ^= 1;
  }

  // C layout: element r of lane -> row = half*8 + r, col = l16.
#pragma unroll
  for (int mi = 0; mi < 4; ++mi)
#pragma unroll
    for (int ni = 0; ni < 2; ++ni)
#pragma unroll
      for (int r = 0; r < 8; ++r) {
        const size_t row = (size_t)(m0 + wm * 64 + mi * 16 + half * 8 + r);
        const size_t col = (size_t)(n0 + wn * 32 + ni * 16 + l16);
        C[row * N + col] = acc[mi][ni][r];
      }
}

// ---------------------------------------------------------------------------
// RoPE: rotate pairs (2i, 2i+1) within each head dim, fp32 -> bf16.
// Works for Q (nheads=32, row=2048) and K (nheads=8, row=512).
// ---------------------------------------------------------------------------
__global__ void rope_kernel(const float* __restrict__ x,
                            const float* __restrict__ cosT,
                            const float* __restrict__ sinT,
                            bf16* __restrict__ out, int nheads, int rowlen, int total) {
  int idx = blockIdx.x * 256 + threadIdx.x;   // over S * nheads * 32
  if (idx >= total) return;
  int i = idx & 31;
  int h = (idx >> 5) % nheads;
  int s = idx / (32 * nheads);
  size_t base = (size_t)s * rowlen + h * 64 + 2 * i;
  float x0 = x[base], x1 = x[base + 1];
  float c  = cosT[s * 32 + i];
  float sn = sinT[s * 32 + i];
  out[base]     = (bf16)(x0 * c - x1 * sn);
  out[base + 1] = (bf16)(x0 * sn + x1 * c);
}

// ---------------------------------------------------------------------------
// V: fp32 [s][kvh*64+d] -> bf16 transposed [kvh*64+d][s]
// ---------------------------------------------------------------------------
__global__ void conv_vt(const float* __restrict__ v, bf16* __restrict__ vt, int S, int total) {
  int idx = blockIdx.x * 256 + threadIdx.x;   // over S * 512
  if (idx >= total) return;
  int col = idx & 511;
  int s   = idx >> 9;
  vt[(size_t)col * S + s] = (bf16)v[(size_t)s * 512 + col];
}

// ---------------------------------------------------------------------------
// Flash attention (no mask). Grid: (S/128, H). 256 threads = 8 waves.
// Each wave owns 16 query rows; online softmax over 32-key blocks.
// QK^T: 4 WMMA per block; P.V: 4 WMMA per block (P transposed via LDS).
// Prefetch keeps the K/V stream ~128 keys ahead.
// ---------------------------------------------------------------------------
__global__ __launch_bounds__(256)
void attn_kernel(const bf16* __restrict__ qb,   // [S][2048]  (rope'd)
                 const bf16* __restrict__ kb,   // [S][512]   (rope'd)
                 const bf16* __restrict__ vt,   // [512][S]   (transposed)
                 bf16* __restrict__ ob,         // [S][2048]
                 int S) {
  __shared__ bf16 pLds[8][16 * 32];             // per-wave P scratch

  const int lane = threadIdx.x & 31;
  const int wave = threadIdx.x >> 5;
  const int l16  = lane & 15;
  const int half = lane >> 4;
  const int h    = blockIdx.y;
  const int hk   = h >> 2;                      // N_REP = 4
  const int q0   = blockIdx.x * 128 + wave * 16;

  // Q fragments for this wave's 16 rows: two K-steps (d = 0..31, 32..63).
  v16bf qF[2];
#pragma unroll
  for (int t = 0; t < 2; ++t) {
    AFrag f;
    const bf16* qrow = qb + (size_t)(q0 + l16) * 2048 + h * 64 + t * 32;
    f.h[0] = *(const v8bf*)(qrow + half * 16);
    f.h[1] = *(const v8bf*)(qrow + half * 16 + 8);
    qF[t] = f.v;
  }

  float mstate[8], lstate[8];
#pragma unroll
  for (int r = 0; r < 8; ++r) { mstate[r] = -1.0e30f; lstate[r] = 0.0f; }
  v8f O[4];
#pragma unroll
  for (int nt = 0; nt < 4; ++nt) O[nt] = (v8f){};

  const float sc = 0.125f;                      // 1/sqrt(64)

  for (int k0 = 0; k0 < S; k0 += 32) {
    // ---- keep the K/V streams ahead of the math (global_prefetch_b8) ----
    if (k0 + 128 < S) {
      __builtin_prefetch(kb + (size_t)(k0 + 128 + lane) * 512 + hk * 64, 0, 1);
      __builtin_prefetch(vt + (size_t)(hk * 64 + lane) * S + k0 + 128, 0, 1);
    }

    // ---- scores: 16 queries x 32 keys ----
    v8f s0 = (v8f){}, s1 = (v8f){};
#pragma unroll
    for (int t = 0; t < 2; ++t) {
      // B-frag for keys : element e -> d = t*32 + half*16 + e.
      v16bf b0 = *(const v16bf*)(kb + (size_t)(k0 + l16) * 512 + hk * 64 + t * 32 + half * 16);
      s0 = wmma_bf16(qF[t], b0, s0);
      v16bf b1 = *(const v16bf*)(kb + (size_t)(k0 + 16 + l16) * 512 + hk * 64 + t * 32 + half * 16);
      s1 = wmma_bf16(qF[t], b1, s1);
    }

    // ---- online softmax over this 32-key block ----
    float rmax[8];
#pragma unroll
    for (int r = 0; r < 8; ++r) {
      float a = fmaxf(s0[r], s1[r]) * sc;
#pragma unroll
      for (int m = 1; m < 16; m <<= 1) a = fmaxf(a, __shfl_xor(a, m, 32));
      rmax[r] = a;
    }
    float alpha[8];
#pragma unroll
    for (int r = 0; r < 8; ++r) {
      float mn = fmaxf(mstate[r], rmax[r]);
      alpha[r] = __expf(mstate[r] - mn);
      mstate[r] = mn;
    }
    v8f p0, p1;
#pragma unroll
    for (int r = 0; r < 8; ++r) {
      p0[r] = __expf(s0[r] * sc - mstate[r]);
      p1[r] = __expf(s1[r] * sc - mstate[r]);
    }
#pragma unroll
    for (int r = 0; r < 8; ++r) {
      float sm = p0[r] + p1[r];
#pragma unroll
      for (int m = 1; m < 16; m <<= 1) sm += __shfl_xor(sm, m, 32);
      lstate[r] = lstate[r] * alpha[r] + sm;
    }
#pragma unroll
    for (int nt = 0; nt < 4; ++nt)
#pragma unroll
      for (int r = 0; r < 8; ++r) O[nt][r] *= alpha[r];

    // ---- transpose P (C-layout -> A-layout) through per-wave LDS ----
    bf16* pw = &pLds[wave][0];
#pragma unroll
    for (int r = 0; r < 8; ++r) {
      const int row = half * 8 + r;
      pw[row * 32 + l16]      = (bf16)p0[r];
      pw[row * 32 + 16 + l16] = (bf16)p1[r];
    }
    AFrag pf;
    pf.h[0] = *(const v8bf*)&pw[l16 * 32 + half * 16];
    pf.h[1] = *(const v8bf*)&pw[l16 * 32 + half * 16 + 8];

    // ---- P(16x32) . V(32x64) : B-frag element e -> key = k0 + half*16 + e ----
#pragma unroll
    for (int nt = 0; nt < 4; ++nt) {
      v16bf bv = *(const v16bf*)(vt + (size_t)(hk * 64 + nt * 16 + l16) * S + k0 + half * 16);
      O[nt] = wmma_bf16(pf.v, bv, O[nt]);
    }
  }

  // ---- epilogue: O / l, write bf16 [s][h*64+d] ----
#pragma unroll
  for (int nt = 0; nt < 4; ++nt)
#pragma unroll
    for (int r = 0; r < 8; ++r) {
      const float val = O[nt][r] / lstate[r];
      const size_t row = (size_t)(q0 + half * 8 + r);
      ob[row * 2048 + h * 64 + nt * 16 + l16] = (bf16)val;
    }
}

// ---------------------------------------------------------------------------
// Host launcher
// ---------------------------------------------------------------------------
extern "C" void kernel_launch(void* const* d_in, const int* in_sizes, int n_in,
                              void* d_out, int out_size, void* d_ws, size_t ws_size,
                              hipStream_t stream) {
  (void)in_sizes; (void)n_in; (void)out_size; (void)ws_size;
  constexpr int S    = 4096;
  constexpr int HID2 = 4096;   // 2*HID (input feature dim)
  constexpr int HD   = 2048;   // H*D
  constexpr int KVD  = 512;    // KVH*D

  const float* hs  = (const float*)d_in[0];
  const float* wq  = (const float*)d_in[1];
  const float* wk  = (const float*)d_in[2];
  const float* wv  = (const float*)d_in[3];
  const float* wo  = (const float*)d_in[4];
  const float* cosT = (const float*)d_in[5];
  const float* sinT = (const float*)d_in[6];
  float* out = (float*)d_out;

  char* ws = (char*)d_ws;
  size_t off = 0;
  auto alloc = [&](size_t bytes) -> void* {
    void* p = ws + off;
    off += (bytes + 255) & ~(size_t)255;
    return p;
  };

  bf16* hb   = (bf16*)alloc((size_t)S * HID2 * 2);
  bf16* wqb  = (bf16*)alloc((size_t)HD * HID2 * 2);
  bf16* wkb  = (bf16*)alloc((size_t)KVD * HID2 * 2);
  bf16* wvb  = (bf16*)alloc((size_t)KVD * HID2 * 2);
  bf16* wob  = (bf16*)alloc((size_t)HD * HD * 2);
  float* qf  = (float*)alloc((size_t)S * HD * 4);
  float* kf  = (float*)alloc((size_t)S * KVD * 4);
  float* vf  = (float*)alloc((size_t)S * KVD * 4);
  bf16* qbf  = (bf16*)alloc((size_t)S * HD * 2);
  bf16* kbf  = (bf16*)alloc((size_t)S * KVD * 2);
  bf16* vtb  = (bf16*)alloc((size_t)KVD * S * 2);
  bf16* attb = (bf16*)alloc((size_t)S * HD * 2);

  auto cgrid = [](int n) { return (n + 255) / 256; };

  // 1) fp32 -> bf16 casts
  cast_f32_bf16<<<cgrid(S * HID2), 256, 0, stream>>>(hs, hb, S * HID2);
  cast_f32_bf16<<<cgrid(HD * HID2), 256, 0, stream>>>(wq, wqb, HD * HID2);
  cast_f32_bf16<<<cgrid(KVD * HID2), 256, 0, stream>>>(wk, wkb, KVD * HID2);
  cast_f32_bf16<<<cgrid(KVD * HID2), 256, 0, stream>>>(wv, wvb, KVD * HID2);
  cast_f32_bf16<<<cgrid(HD * HD), 256, 0, stream>>>(wo, wob, HD * HD);

  // 2) QKV projections (WMMA GEMM)
  gemm_bf16<<<dim3(S / GT_M, HD / GT_N), 256, 0, stream>>>(hb, wqb, qf, S, HD, HID2);
  gemm_bf16<<<dim3(S / GT_M, KVD / GT_N), 256, 0, stream>>>(hb, wkb, kf, S, KVD, HID2);
  gemm_bf16<<<dim3(S / GT_M, KVD / GT_N), 256, 0, stream>>>(hb, wvb, vf, S, KVD, HID2);

  // 3) RoPE + V transpose, emit bf16
  rope_kernel<<<cgrid(S * 32 * 32), 256, 0, stream>>>(qf, cosT, sinT, qbf, 32, HD, S * 32 * 32);
  rope_kernel<<<cgrid(S * 8 * 32), 256, 0, stream>>>(kf, cosT, sinT, kbf, 8, KVD, S * 8 * 32);
  conv_vt<<<cgrid(S * KVD), 256, 0, stream>>>(vf, vtb, S, S * KVD);

  // 4) attention (WMMA flash attention, no mask)
  attn_kernel<<<dim3(S / 128, 32), 256, 0, stream>>>(qbf, kbf, vtb, attb, S);

  // 5) output projection (WMMA GEMM) -> fp32 d_out
  gemm_bf16<<<dim3(S / GT_M, HD / GT_N), 256, 0, stream>>>(attb, wob, out, S, HD, HD);
}
